// FineGrainedGate_21294447854271
// MI455X (gfx1250) — matrix operand order repaired and checked by reading, not compiled, not run on hardware
//
#include <hip/hip_runtime.h>

#define DEVI __device__ __forceinline__

typedef __attribute__((ext_vector_type(16))) __bf16 v16bf;
typedef __attribute__((ext_vector_type(8)))  float  v8f;

union BFrag { v16bf v; unsigned u[8]; uint4 q[2]; };
union U4S   { uint4 q; unsigned short s[8]; };

static constexpr int Bb = 2, T = 2048, Rr = 1024, Dd = 1024, Hh = 16, HD = 64;

DEVI unsigned short f2bf(float f) {
  unsigned u = __float_as_uint(f);
  u += 0x7fffu + ((u >> 16) & 1u);      // RNE fp32 -> bf16
  return (unsigned short)(u >> 16);
}
DEVI unsigned pack2(float a, float b) {
  return (unsigned)f2bf(a) | ((unsigned)f2bf(b) << 16);
}
DEVI float wmaxf(float v) { for (int o = 16; o; o >>= 1) v = fmaxf(v, __shfl_xor(v, o, 32)); return v; }
DEVI float wsumf(float v) { for (int o = 16; o; o >>= 1) v += __shfl_xor(v, o, 32); return v; }

DEVI v8f wmma_bf16(const BFrag& a, const BFrag& b, v8f c) {
  return __builtin_amdgcn_wmma_f32_16x16x32_bf16(false, a.v, false, b.v, (short)0, c, false, false);
}
DEVI unsigned lds_off(const void* p) {        // LDS aperture: offset lives in addr[31:0]
  return (unsigned)(unsigned long long)p;
}
DEVI void async_ld_b128(unsigned lds, unsigned goff, const void* base) {
  asm volatile("global_load_async_to_lds_b128 %0, %1, %2"
               :: "v"(lds), "v"(goff), "s"(base) : "memory");
}
DEVI void wait_async0() { asm volatile("s_wait_asynccnt 0x0" ::: "memory"); }

// ---------------- elementwise fp32 -> bf16 (8 elems/thread) ----------------
__global__ __launch_bounds__(256)
void f32_to_bf16(const float* __restrict__ x, unsigned short* __restrict__ y, unsigned n) {
  unsigned i = (blockIdx.x * 256u + threadIdx.x) * 8u;
  if (i >= n) return;
  float4 a = *(const float4*)(x + i);
  float4 b = *(const float4*)(x + i + 4);
  uint4 o;
  o.x = pack2(a.x, a.y); o.y = pack2(a.z, a.w);
  o.z = pack2(b.x, b.y); o.w = pack2(b.z, b.w);
  *(uint4*)(y + i) = o;
}

// ---------------- LayerNorm rows -> bf16 (one wave per row, D=1024) ----------------
__global__ __launch_bounds__(256)
void ln_rows(const float* __restrict__ X, const float* __restrict__ g,
             const float* __restrict__ be, unsigned short* __restrict__ Y, int nrows) {
  const int w = threadIdx.x >> 5, lane = threadIdx.x & 31;
  const int row = blockIdx.x * 8 + w;
  if (row >= nrows) return;
  const float* x = X + (size_t)row * Dd;
  float v[32]; float s = 0.f;
#pragma unroll
  for (int j = 0; j < 32; j++) { v[j] = x[lane + 32 * j]; s += v[j]; }
  const float mean = wsumf(s) * (1.0f / Dd);
  float q = 0.f;
#pragma unroll
  for (int j = 0; j < 32; j++) { float d = v[j] - mean; q += d * d; }
  const float inv = rsqrtf(wsumf(q) * (1.0f / Dd) + 1e-5f);
  unsigned short* y = Y + (size_t)row * Dd;
#pragma unroll
  for (int j = 0; j < 32; j++) {
    int i = lane + 32 * j;
    y[i] = f2bf((v[j] - mean) * inv * g[i] + be[i]);
  }
}

// --------- gate tail: r = sigmoid(Hg[row,0:512] . w2 + b2) (wave/row) ---------
__global__ __launch_bounds__(256)
void gate_reduce(const float* __restrict__ Hg, const float* __restrict__ w2,
                 const float* __restrict__ b2, float* __restrict__ r, int nrows) {
  const int w = threadIdx.x >> 5, lane = threadIdx.x & 31;
  const int row = blockIdx.x * 8 + w;
  if (row >= nrows) return;
  const float* h = Hg + (size_t)row * 512;
  float s = 0.f;
#pragma unroll
  for (int j = 0; j < 16; j++) { int i = lane + 32 * j; s += h[i] * w2[i]; }
  s = wsumf(s);
  if (lane == 0) r[row] = 1.f / (1.f + expf(-(s + b2[0])));
}

// ---------------- WMMA GEMM on bf16 operands: C = X[N,Kd] @ W[M,Kd]^T ----------------
// 128x128 WG tile, 64-deep k stage, double-buffered global_load_async_to_lds_b128 staging
// (ASYNCcnt) so DMA of stage s+1 overlaps the 16 WMMAs of stage s.
// mode 1: bf16 store (QKV)  mode 2: out = Ht + tanh(gamma)*rt[row]*C  mode 3: fp32 GELU(C+bias)
__global__ __launch_bounds__(256)
void gemm_bf16(const unsigned short* __restrict__ Xb, const unsigned short* __restrict__ Wb,
               const float* __restrict__ bias, int Kd, int Mtot, int mode,
               float* __restrict__ Cf, unsigned short* __restrict__ Cb,
               const float* __restrict__ Ht, const float* __restrict__ rt,
               const float* __restrict__ gamma) {
  constexpr int AP = 72;                         // 64 bf16 + 8 pad (144 B pitch, b128 conflict-free)
  __shared__ unsigned short As[2][128 * AP];     // 2 x 18 KB
  __shared__ unsigned short Bs[2][128 * AP];     // 2 x 18 KB
  const int tid = threadIdx.x, w = tid >> 5, lane = tid & 31;
  const int rowBase = blockIdx.y * 128, colBase = blockIdx.x * 128;
  const int wr = w & 3, wc = w >> 2;             // wave tile: 32 rows x 64 cols
  const unsigned asB = lds_off(As[0]), bsB = lds_off(Bs[0]);
  constexpr unsigned BUFB = 128u * AP * 2u;      // bytes per buffer

  v8f acc[2][4];
#pragma unroll
  for (int a = 0; a < 2; a++)
#pragma unroll
    for (int b = 0; b < 4; b++) acc[a][b] = (v8f){0.f,0.f,0.f,0.f,0.f,0.f,0.f,0.f};

  const int rr0 = tid >> 3;                      // 0..31
  const int cc  = (tid & 7) * 8;                 // bf16 col within 64-wide stage
  auto issue = [&](int kc, int buf) {
#pragma unroll
    for (int p = 0; p < 4; p++) {                // A: 128 rows x 64 bf16
      int rr = rr0 + p * 32;
      async_ld_b128(asB + buf * BUFB + (unsigned)(rr * AP + cc) * 2,
                    (unsigned)(((size_t)(rowBase + rr) * Kd + kc + cc) * 2), Xb);
      async_ld_b128(bsB + buf * BUFB + (unsigned)(rr * AP + cc) * 2,
                    (unsigned)(((size_t)(colBase + rr) * Kd + kc + cc) * 2), Wb);
    }
  };

  const int S = Kd / 64;
  issue(0, 0);
  for (int s = 0; s < S; s++) {
    wait_async0();                               // own stage-s async writes done
    __syncthreads();                             // all waves' writes done; prev-buffer reads done
    if (s + 1 < S) issue((s + 1) * 64, (s + 1) & 1);   // DMA overlaps compute below

    const unsigned short* Ab = As[s & 1];
    const unsigned short* Bp = Bs[s & 1];
    const int am = lane & 15;
    const int kb1 = (lane >> 4) * 8;             // A k-groups {0..7,16..23} / {8..15,24..31}
    const int kbB = (lane >> 4) * 8;             // B k-groups {0..7,16..23} / {8..15,24..31}
#pragma unroll
    for (int c = 0; c < 2; c++) {                // two 32-deep chunks of the stage
      BFrag a[2], bf[4];
#pragma unroll
      for (int t = 0; t < 2; t++) {
        const unsigned short* pa = Ab + (wr * 32 + t * 16 + am) * AP + c * 32 + kb1;
        a[t].q[0] = *(const uint4*)pa;
        a[t].q[1] = *(const uint4*)(pa + 16);
      }
#pragma unroll
      for (int t = 0; t < 4; t++) {
        const unsigned short* pb = Bp + (wc * 64 + t * 16 + am) * AP + c * 32 + kbB;
        bf[t].q[0] = *(const uint4*)pb;
        bf[t].q[1] = *(const uint4*)(pb + 16);
      }
#pragma unroll
      for (int tm = 0; tm < 2; tm++)
#pragma unroll
        for (int tn = 0; tn < 4; tn++)
          acc[tm][tn] = wmma_bf16(a[tm], bf[tn], acc[tm][tn]);
    }
  }

  const float tg = (mode == 2) ? tanhf(gamma[0]) : 0.f;
#pragma unroll
  for (int tm = 0; tm < 2; tm++)
#pragma unroll
    for (int tn = 0; tn < 4; tn++)
#pragma unroll
      for (int i = 0; i < 8; i++) {
        int row = rowBase + wr * 32 + tm * 16 + i + ((lane >> 4) << 3);
        int col = colBase + wc * 64 + tn * 16 + (lane & 15);
        float val = acc[tm][tn][i];
        size_t off = (size_t)row * Mtot + col;
        if (mode == 1) {
          Cb[off] = f2bf(val);
        } else if (mode == 2) {
          Cf[off] = Ht[off] + tg * rt[row] * val;
        } else {
          float z = val + bias[col];
          Cf[off] = 0.5f * z * (1.f + erff(z * 0.70710678118f));  // exact GELU
        }
      }
}

// ---------------- fused attention: one WG per (b, h, 16-token tile) ----------------
__global__ __launch_bounds__(256)
void attn_kernel(const unsigned short* __restrict__ Qb,
                 const unsigned short* __restrict__ Kb,
                 const unsigned short* __restrict__ Vb,
                 const int* __restrict__ maskv,
                 const float* __restrict__ r_t, const float* __restrict__ r_v,
                 unsigned short* __restrict__ ctxb) {
  constexpr int LP = Rr + 8;     // logits/W row pitch (floats)
  constexpr int VP = 512 + 8;    // transposed-V pitch (ushorts), half of R staged
  __shared__ float Ls[16 * LP];              // 66 KB: logits, then final weights W
  __shared__ unsigned short Vt[64 * VP];     // 65 KB: V^T half (d-major)
  __shared__ float cpart[4 * 16 * 16];       // cross-wave ctx partials
  const int tid = threadIdx.x, w = tid >> 5, lane = tid & 31;
  const int tb = blockIdx.x, h = blockIdx.y, b = blockIdx.z;
  const int t0 = tb * 16;
  const float NEGI = -3.4e38f;

  auto stageV = [&](int half) {
#pragma unroll 4
    for (int p = 0; p < 16; p++) {
      int r = (tid >> 3) + p * 32;                 // 0..511 (local)
      int db = (tid & 7) * 8;
      U4S u;
      u.q = *(const uint4*)(Vb + (size_t)(b * Rr + half * 512 + r) * Dd + h * HD + db);
#pragma unroll
      for (int j = 0; j < 8; j++) Vt[(db + j) * VP + r] = u.s[j];
    }
  };

  // Q fragments (2 k-chunks over HD=64), kept in regs all kernel
  BFrag aq[2];
  const size_t qrow = (size_t)(b * T + t0 + (lane & 15)) * Dd + h * HD;
#pragma unroll
  for (int c = 0; c < 2; c++) {
    int kb = c * 32 + (lane >> 4) * 8;
    aq[c].q[0] = *(const uint4*)(Qb + qrow + kb);
    aq[c].q[1] = *(const uint4*)(Qb + qrow + kb + 16);
  }

  stageV(0);

  // ---- logits = Q K^T / 8 for all R (K streamed from L2, prefetch next tile) ----
  for (int ii = 0; ii < 8; ii++) {
    int r0 = (w + ii * 8) * 16;
    v8f acc = (v8f){0.f,0.f,0.f,0.f,0.f,0.f,0.f,0.f};
    const size_t krow = (size_t)(b * Rr + r0 + (lane & 15)) * Dd + h * HD;
    if (ii < 7) {
      const size_t knext = (size_t)(b * Rr + (w + (ii + 1) * 8) * 16 + (lane & 15)) * Dd + h * HD;
      __builtin_prefetch(Kb + knext, 0, 0);        // global_prefetch_b8
    }
#pragma unroll
    for (int c = 0; c < 2; c++) {
      BFrag bk;
      int kb = c * 32 + (lane >> 4) * 8;
      bk.q[0] = *(const uint4*)(Kb + krow + kb);
      bk.q[1] = *(const uint4*)(Kb + krow + kb + 16);
      acc = wmma_bf16(aq[c], bk, acc);
    }
    int n = r0 + (lane & 15);
#pragma unroll
    for (int i = 0; i < 8; i++) {
      int m = i + ((lane >> 4) << 3);
      Ls[m * LP + n] = acc[i] * 0.125f;            // 1/sqrt(64)
    }
  }
  __syncthreads();

  // ---- per-row: masked softmax A, top-32 threshold, sparse softmax G, gate mix ----
  for (int rr = 0; rr < 2; rr++) {
    const int m = w * 2 + rr;
    float l[32]; unsigned mkb = 0;
#pragma unroll
    for (int j = 0; j < 32; j++) {
      int idx = lane + 32 * j;
      l[j] = Ls[m * LP + idx];
      if (maskv[b * Rr + idx]) mkb |= (1u << j);
    }
    float vmax = NEGI;
#pragma unroll
    for (int j = 0; j < 32; j++) if ((mkb >> j) & 1u) vmax = fmaxf(vmax, l[j]);
    vmax = wmaxf(vmax);
    float s = 0.f;
#pragma unroll
    for (int j = 0; j < 32; j++) if ((mkb >> j) & 1u) s += expf(l[j] - vmax);
    s = wsumf(s);
    const float inv_s = (s > 0.f) ? 1.f / s : 0.f;

    // 32 iterative wave-argmax extractions -> 32nd-largest threshold
    unsigned rm = 0;
    float thr = NEGI, gmax = NEGI;
    for (int it = 0; it < 32; it++) {
      float lm = NEGI; int jm = 0;
#pragma unroll
      for (int j = 0; j < 32; j++)
        if (!((rm >> j) & 1u) && l[j] > lm) { lm = l[j]; jm = j; }
      float v = lm; int li = lane;
#pragma unroll
      for (int o = 16; o; o >>= 1) {
        float ov = __shfl_xor(v, o, 32);
        int   ol = __shfl_xor(li, o, 32);
        if (ov > v || (ov == v && ol < li)) { v = ov; li = ol; }
      }
      if (it == 0) gmax = v;
      thr = v;
      if (lane == li) rm |= (1u << jm);
    }
    float gs = 0.f;
#pragma unroll
    for (int j = 0; j < 32; j++) if (l[j] >= thr) gs += expf(l[j] - gmax);
    gs = wsumf(gs);
    const float inv_g = 1.f / gs;
    const float rt = r_t[b * T + t0 + m];
#pragma unroll
    for (int j = 0; j < 32; j++) {
      int idx = lane + 32 * j;
      float a = ((mkb >> j) & 1u) ? expf(l[j] - vmax) * inv_s : 0.f;
      float g = (l[j] >= thr) ? expf(l[j] - gmax) * inv_g : 0.f;
      float rv = r_v[b * Rr + idx];
      Ls[m * LP + idx] = 0.35f * (a + g) + 0.3f * rt * rv;   // (1-beta)*mix + beta*outer
    }
  }
  __syncthreads();

  // ---- ctx = W @ V via WMMA, V^T in LDS, 2 halves of R ----
  const int ci = w & 3, sub = w >> 2;
  const int d0 = ci * 16;
  const int am = lane & 15, g1 = (lane >> 4) * 8;
  v8f acc = (v8f){0.f,0.f,0.f,0.f,0.f,0.f,0.f,0.f};
  for (int half = 0; half < 2; half++) {
    if (half == 1) {
      __syncthreads();          // all reads of Vt(half 0) done
      stageV(1);
      __syncthreads();
    }
#pragma unroll
    for (int j = 0; j < 8; j++) {
      int rb = half * 512 + sub * 256 + j * 32;  // global r base of this 32-chunk
      int rl = rb - half * 512;                  // local base in Vt
      BFrag a;                                   // W rows (fp32 in LDS) -> bf16
      const float* rowp = Ls + am * LP + rb + g1;
      float4 f0 = *(const float4*)(rowp);
      float4 f1 = *(const float4*)(rowp + 4);
      float4 f2 = *(const float4*)(rowp + 16);
      float4 f3 = *(const float4*)(rowp + 20);
      a.u[0] = pack2(f0.x, f0.y); a.u[1] = pack2(f0.z, f0.w);
      a.u[2] = pack2(f1.x, f1.y); a.u[3] = pack2(f1.z, f1.w);
      a.u[4] = pack2(f2.x, f2.y); a.u[5] = pack2(f2.z, f2.w);
      a.u[6] = pack2(f3.x, f3.y); a.u[7] = pack2(f3.z, f3.w);
      BFrag bv;                                  // V^T column n=d contiguous over r
      const unsigned short* pv = Vt + (d0 + am) * VP + rl + (lane >> 4) * 8;
      bv.q[0] = *(const uint4*)pv;
      bv.q[1] = *(const uint4*)(pv + 16);
      acc = wmma_bf16(a, bv, acc);
    }
  }
  if (w >= 4) {
#pragma unroll
    for (int i = 0; i < 8; i++) {
      int m = i + ((lane >> 4) << 3), n = lane & 15;
      cpart[(ci * 16 + m) * 16 + n] = acc[i];
    }
  }
  __syncthreads();
  if (w < 4) {
#pragma unroll
    for (int i = 0; i < 8; i++) {
      int m = i + ((lane >> 4) << 3), n = lane & 15;
      float v = acc[i] + cpart[(ci * 16 + m) * 16 + n];
      ctxb[(size_t)(b * T + t0 + m) * Dd + h * HD + d0 + n] = f2bf(v);
    }
  }
}

// ------------------------------- host orchestration -------------------------------
extern "C" void kernel_launch(void* const* d_in, const int* in_sizes, int n_in,
                              void* d_out, int out_size, void* d_ws, size_t ws_size,
                              hipStream_t stream) {
  (void)in_sizes; (void)n_in; (void)out_size; (void)ws_size;
  const float* H_t     = (const float*)d_in[0];
  const float* H_v     = (const float*)d_in[1];
  const int*   maskv   = (const int*)d_in[2];
  const float* Wq      = (const float*)d_in[3];
  const float* Wk      = (const float*)d_in[4];
  const float* Wv      = (const float*)d_in[5];
  const float* gt_ln_g = (const float*)d_in[6];
  const float* gt_ln_b = (const float*)d_in[7];
  const float* gt_w1   = (const float*)d_in[8];
  const float* gt_b1   = (const float*)d_in[9];
  const float* gt_w2   = (const float*)d_in[10];
  const float* gt_b2   = (const float*)d_in[11];
  const float* gv_ln_g = (const float*)d_in[12];
  const float* gv_ln_b = (const float*)d_in[13];
  const float* gv_w1   = (const float*)d_in[14];
  const float* gv_b1   = (const float*)d_in[15];
  const float* gv_w2   = (const float*)d_in[16];
  const float* gv_b2   = (const float*)d_in[17];
  const float* Wo      = (const float*)d_in[18];
  const float* gamma   = (const float*)d_in[19];
  float* out = (float*)d_out;

  char* ws = (char*)d_ws;
  size_t o = 0;
  auto alloc = [&](size_t bytes) -> void* {
    void* p = ws + o;
    o += (bytes + 255) & ~(size_t)255;
    return p;
  };
  const int NT = Bb * T;    // 4096
  const int NR = Bb * Rr;   // 2048
  unsigned short* Htb = (unsigned short*)alloc((size_t)NT * Dd * 2);
  unsigned short* Hvb = (unsigned short*)alloc((size_t)NR * Dd * 2);
  unsigned short* Xtb = (unsigned short*)alloc((size_t)NT * Dd * 2);   // LN(H_t) bf16
  unsigned short* Xvb = (unsigned short*)alloc((size_t)NR * Dd * 2);
  float* Hgt = (float*)alloc((size_t)NT * 512 * 4);
  float* Hgv = (float*)alloc((size_t)NR * 512 * 4);
  float* rt  = (float*)alloc((size_t)NT * 4);
  float* rv  = (float*)alloc((size_t)NR * 4);
  unsigned short* Qb  = (unsigned short*)alloc((size_t)NT * Dd * 2);
  unsigned short* Kb  = (unsigned short*)alloc((size_t)NR * Dd * 2);
  unsigned short* Vb  = (unsigned short*)alloc((size_t)NR * Dd * 2);
  unsigned short* ctb = (unsigned short*)alloc((size_t)NT * Dd * 2);   // ctx bf16
  unsigned short* Wqb = (unsigned short*)alloc((size_t)Dd * Dd * 2);
  unsigned short* Wkb = (unsigned short*)alloc((size_t)Dd * Dd * 2);
  unsigned short* Wvb = (unsigned short*)alloc((size_t)Dd * Dd * 2);
  unsigned short* Wob = (unsigned short*)alloc((size_t)Dd * Dd * 2);
  unsigned short* W1t = (unsigned short*)alloc((size_t)512 * Dd * 2);
  unsigned short* W1v = (unsigned short*)alloc((size_t)512 * Dd * 2);

  const dim3 blk(256);
  const unsigned nAct = (unsigned)NT * Dd, nActV = (unsigned)NR * Dd;
  const unsigned nW = (unsigned)Dd * Dd, nW1 = 512u * Dd;

  // bf16 pre-conversion (amortizes per-tile conversion across all GEMM reuse)
  f32_to_bf16<<<nAct / 2048, blk, 0, stream>>>(H_t, Htb, nAct);
  f32_to_bf16<<<nActV / 2048, blk, 0, stream>>>(H_v, Hvb, nActV);
  f32_to_bf16<<<nW / 2048, blk, 0, stream>>>(Wq, Wqb, nW);
  f32_to_bf16<<<nW / 2048, blk, 0, stream>>>(Wk, Wkb, nW);
  f32_to_bf16<<<nW / 2048, blk, 0, stream>>>(Wv, Wvb, nW);
  f32_to_bf16<<<nW / 2048, blk, 0, stream>>>(Wo, Wob, nW);
  f32_to_bf16<<<nW1 / 2048, blk, 0, stream>>>(gt_w1, W1t, nW1);
  f32_to_bf16<<<nW1 / 2048, blk, 0, stream>>>(gv_w1, W1v, nW1);

  // gate branches
  ln_rows<<<NT / 8, blk, 0, stream>>>(H_t, gt_ln_g, gt_ln_b, Xtb, NT);
  ln_rows<<<NR / 8, blk, 0, stream>>>(H_v, gv_ln_g, gv_ln_b, Xvb, NR);
  gemm_bf16<<<dim3(512 / 128, NT / 128), blk, 0, stream>>>(
      Xtb, W1t, gt_b1, Dd, 512, 3, Hgt, nullptr, nullptr, nullptr, nullptr);
  gemm_bf16<<<dim3(512 / 128, NR / 128), blk, 0, stream>>>(
      Xvb, W1v, gv_b1, Dd, 512, 3, Hgv, nullptr, nullptr, nullptr, nullptr);
  gate_reduce<<<NT / 8, blk, 0, stream>>>(Hgt, gt_w2, gt_b2, rt, NT);
  gate_reduce<<<NR / 8, blk, 0, stream>>>(Hgv, gv_w2, gv_b2, rv, NR);

  // projections (bf16 outputs for WMMA attention)
  gemm_bf16<<<dim3(Dd / 128, NT / 128), blk, 0, stream>>>(
      Htb, Wqb, nullptr, Dd, Dd, 1, nullptr, Qb, nullptr, nullptr, nullptr);
  gemm_bf16<<<dim3(Dd / 128, NR / 128), blk, 0, stream>>>(
      Hvb, Wkb, nullptr, Dd, Dd, 1, nullptr, Kb, nullptr, nullptr, nullptr);
  gemm_bf16<<<dim3(Dd / 128, NR / 128), blk, 0, stream>>>(
      Hvb, Wvb, nullptr, Dd, Dd, 1, nullptr, Vb, nullptr, nullptr, nullptr);

  // fused attention (softmax + top-k + gates + ctx)
  attn_kernel<<<dim3(T / 16, Hh, Bb), blk, 0, stream>>>(Qb, Kb, Vb, maskv, rt, rv, ctb);

  // output projection with fused residual epilogue
  gemm_bf16<<<dim3(Dd / 128, NT / 128), blk, 0, stream>>>(
      ctb, Wob, nullptr, Dd, Dd, 2, out, nullptr, H_t, rt, gamma);
}
